// DockingScorePredictor_50912542326916
// MI455X (gfx1250) — matrix operand order repaired and checked by reading, not compiled, not run on hardware
//
#include <hip/hip_runtime.h>
#include <hip/hip_bf16.h>

// ---------------------------------------------------------------------------
// Problem constants (match reference)
// ---------------------------------------------------------------------------
#define NB   8      // complexes
#define NP   512    // protein atoms / complex
#define NL   64     // ligand atoms / complex
#define NH   128    // hidden dim
#define NRB  32     // radial bins
#define CUT  8.0f

typedef __attribute__((ext_vector_type(16))) __bf16 v16bf;
typedef __attribute__((ext_vector_type(8)))  float  v8f;

union FragU { v16bf v; uint4 q[2]; };

__device__ __forceinline__ unsigned short f2bf(float x) {
    unsigned u = __float_as_uint(x);
    unsigned r = u + 0x7FFFu + ((u >> 16) & 1u);   // round-to-nearest-even
    return (unsigned short)(r >> 16);
}

// A-fragment (16x32 bf16) from row-major LDS: lane = row (lanes 0-15 / 16-31
// mirror rows), per-lane K = {16*half .. +15}
__device__ __forceinline__ v16bf afrag_ld(const unsigned short* base, int row0,
                                          int strideElems, int kOff, int lane) {
    const int row  = lane & 15;
    const int half = lane >> 4;
    const unsigned short* p = base + (row0 + row) * strideElems + kOff + half * 16;
    FragU u;
    u.q[0] = *(const uint4*)(p);
    u.q[1] = *(const uint4*)(p + 8);
    return u.v;
}

// B-fragment (32x16 bf16) from pre-swizzled LDS: fragment-major, 32B/lane
__device__ __forceinline__ v16bf bfrag_ld(const unsigned short* w, int ktile,
                                          int ntile, int lane) {
    const unsigned short* p = w + (((ktile * 8 + ntile) * 32 + lane) << 4);
    FragU u;
    u.q[0] = *(const uint4*)(p);
    u.q[1] = *(const uint4*)(p + 8);
    return u.v;
}

#define WMMA_BF16(A, Bm, C) \
    __builtin_amdgcn_wmma_f32_16x16x32_bf16(false, (A), false, (Bm), (short)0, (C), false, false)

// ---------------------------------------------------------------------------
// Kernel 0: zero the per-complex accumulators in workspace
// ---------------------------------------------------------------------------
__global__ void dock_zero_kernel(float* gsum, float* gcnt) {
    int t = blockIdx.x * blockDim.x + threadIdx.x;
    if (t < NB * NH) gsum[t] = 0.0f;
    if (t < NB)      gcnt[t] = 0.0f;
}

// ---------------------------------------------------------------------------
// Kernel 1: precompute  Abuf[i,:] = b1 + prot_emb[ptype[i]] @ W1[0:H]
//                       Cbuf[j,:] =      lig_emb[ltype[j]]  @ W1[H:2H]
// (0.15 GFLOP total, scalar is fine)
// ---------------------------------------------------------------------------
__global__ __launch_bounds__(NH) void dock_precompute_kernel(
    const float* __restrict__ prot_emb, const float* __restrict__ lig_emb,
    const int* __restrict__ ptype, const int* __restrict__ ltype,
    const float* __restrict__ W1, const float* __restrict__ b1,
    float* __restrict__ Abuf, float* __restrict__ Cbuf) {
    __shared__ float e[NH];
    const int i = blockIdx.x;
    const int n = threadIdx.x;
    if (i < NB * NP) {
        const int ty = ptype[i];
        e[n] = prot_emb[ty * NH + n];
        __syncthreads();
        float acc = b1[n];
        #pragma unroll 8
        for (int k = 0; k < NH; ++k) acc = fmaf(e[k], W1[k * NH + n], acc);
        Abuf[i * NH + n] = acc;
    } else {
        const int j = i - NB * NP;
        const int ty = ltype[j];
        e[n] = lig_emb[ty * NH + n];
        __syncthreads();
        float acc = 0.0f;
        #pragma unroll 8
        for (int k = 0; k < NH; ++k) acc = fmaf(e[k], W1[(NH + k) * NH + n], acc);
        Cbuf[j * NH + n] = acc;
    }
}

// ---------------------------------------------------------------------------
// Kernel 2 (main): per block = (complex b, 2 protein atoms) x 64 ligands
//   -> 128 pair rows, 8 waves, wave w owns row-tile 16w.
//   layer1: rb(K=32) WMMA + Abuf/Cbuf broadcast add, relu
//   layer2/3: K=128 WMMA from LDS-resident swizzled W2/W3
//   layer3 epilogue: mask * h, reduce into per-complex global sums
// ---------------------------------------------------------------------------
__global__ __launch_bounds__(256) void dock_pair_mlp_kernel(
    const float* __restrict__ ppos, const float* __restrict__ lpos,
    const float* __restrict__ W1,  const float* __restrict__ W2,
    const float* __restrict__ b2,  const float* __restrict__ W3,
    const float* __restrict__ b3,  const float* __restrict__ Abuf,
    const float* __restrict__ Cbuf, float* __restrict__ gsum,
    float* __restrict__ gcnt) {
    // LDS: ~147 KB -> 2 workgroups per 320KB WGP
    __shared__ __align__(16) unsigned short w1c[NRB * NH];     //  8 KB swizzled
    __shared__ __align__(16) unsigned short w2s[NH * NH];      // 32 KB swizzled
    __shared__ __align__(16) unsigned short w3s[NH * NH];      // 32 KB swizzled
    __shared__ __align__(16) unsigned short h1[128 * NH];      // 32 KB row-major
    __shared__ __align__(16) unsigned short h2[128 * NH];      // 32 KB row-major
    __shared__ __align__(16) unsigned short rbs[128 * NRB];    //  8 KB row-major
    __shared__ float alds[2 * NH];
    __shared__ float b2l[NH], b3l[NH];
    __shared__ float mlds[128];
    __shared__ float accum[NH];

    const int tid   = threadIdx.x;
    const int bp    = blockIdx.x;
    const int b     = bp >> 8;            // NP/2 = 256 blocks per complex
    const int pbase = (bp & 255) * 2;

    // ---- stage weights into B-fragment swizzled layout -------------------
    for (int idx = tid; idx < NRB * NH; idx += 256) {
        int k = idx >> 7, n = idx & 127;
        int half = (k >> 4) & 1, e = k & 15;
        int dst = ((((k >> 5) * 8 + (n >> 4)) * 32 + half * 16 + (n & 15)) << 4) + e;
        w1c[dst] = f2bf(W1[(2 * NH + k) * NH + n]);
    }
    for (int idx = tid; idx < NH * NH; idx += 256) {
        int k = idx >> 7, n = idx & 127;
        int half = (k >> 4) & 1, e = k & 15;
        int dst = ((((k >> 5) * 8 + (n >> 4)) * 32 + half * 16 + (n & 15)) << 4) + e;
        w2s[dst] = f2bf(W2[idx]);
        w3s[dst] = f2bf(W3[idx]);
    }
    if (tid < NH) { b2l[tid] = b2[tid]; b3l[tid] = b3[tid]; accum[tid] = 0.0f; }
    alds[tid] = Abuf[(b * NP + pbase + (tid >> 7)) * NH + (tid & 127)];

    // ---- distances + radial basis (rows 0..127) --------------------------
    if (tid < 128) {
        const int m = tid;
        const int p = pbase + (m >> 6);
        const int l = m & 63;
        const float dx = ppos[(b * NP + p) * 3 + 0] - lpos[(b * NL + l) * 3 + 0];
        const float dy = ppos[(b * NP + p) * 3 + 1] - lpos[(b * NL + l) * 3 + 1];
        const float dz = ppos[(b * NP + p) * 3 + 2] - lpos[(b * NL + l) * 3 + 2];
        const float dist = sqrtf(dx * dx + dy * dy + dz * dz);
        mlds[m] = (dist < CUT) ? 1.0f : 0.0f;
        const float invw = 1.0f / (0.5f * CUT / NRB + 1e-8f);
        #pragma unroll
        for (int i = 0; i < NRB; ++i) {
            const float c = CUT * (float)i * (1.0f / (NRB - 1));
            const float t = (dist - c) * invw;
            rbs[m * NRB + i] = f2bf(__expf(-0.5f * t * t));
        }
    }
    __syncthreads();

    if (tid == 0) {   // valid-pair count for this block
        float s = 0.0f;
        for (int m = 0; m < 128; ++m) s += mlds[m];
        atomicAdd(&gcnt[b], s);
    }

    const int wv   = tid >> 5;
    const int lane = tid & 31;
    const int row0 = wv * 16;
    const int col  = lane & 15;
    const int half = lane >> 4;

    // ---- layer 1: h1 = relu(A[p] + C[l] + rb @ W1c) ----------------------
    {
        const v16bf arb = afrag_ld(rbs, row0, NRB, 0, lane);
        #pragma unroll
        for (int nt = 0; nt < 8; ++nt) {
            v8f acc = {};
            acc = WMMA_BF16(arb, bfrag_ld(w1c, 0, nt, lane), acc);
            const int n = nt * 16 + col;
            #pragma unroll
            for (int j = 0; j < 8; ++j) {
                const int m = row0 + j + 8 * half;
                float v = acc[j] + alds[(m >> 6) * NH + n]
                                 + Cbuf[(b * NL + (m & 63)) * NH + n];
                h1[m * NH + n] = f2bf(fmaxf(v, 0.0f));
            }
        }
    }

    // ---- layer 2: h2 = relu(h1 @ W2 + b2)  (wave-private rows, no barrier)
    {
        const v16bf a0 = afrag_ld(h1, row0, NH, 0,  lane);
        const v16bf a1 = afrag_ld(h1, row0, NH, 32, lane);
        const v16bf a2 = afrag_ld(h1, row0, NH, 64, lane);
        const v16bf a3 = afrag_ld(h1, row0, NH, 96, lane);
        #pragma unroll
        for (int nt = 0; nt < 8; ++nt) {
            v8f acc = {};
            acc = WMMA_BF16(a0, bfrag_ld(w2s, 0, nt, lane), acc);
            acc = WMMA_BF16(a1, bfrag_ld(w2s, 1, nt, lane), acc);
            acc = WMMA_BF16(a2, bfrag_ld(w2s, 2, nt, lane), acc);
            acc = WMMA_BF16(a3, bfrag_ld(w2s, 3, nt, lane), acc);
            const int n = nt * 16 + col;
            #pragma unroll
            for (int j = 0; j < 8; ++j) {
                const int m = row0 + j + 8 * half;
                h2[m * NH + n] = f2bf(fmaxf(acc[j] + b2l[n], 0.0f));
            }
        }
    }

    // ---- layer 3 + masked reduction --------------------------------------
    {
        const v16bf a0 = afrag_ld(h2, row0, NH, 0,  lane);
        const v16bf a1 = afrag_ld(h2, row0, NH, 32, lane);
        const v16bf a2 = afrag_ld(h2, row0, NH, 64, lane);
        const v16bf a3 = afrag_ld(h2, row0, NH, 96, lane);
        #pragma unroll
        for (int nt = 0; nt < 8; ++nt) {
            v8f acc = {};
            acc = WMMA_BF16(a0, bfrag_ld(w3s, 0, nt, lane), acc);
            acc = WMMA_BF16(a1, bfrag_ld(w3s, 1, nt, lane), acc);
            acc = WMMA_BF16(a2, bfrag_ld(w3s, 2, nt, lane), acc);
            acc = WMMA_BF16(a3, bfrag_ld(w3s, 3, nt, lane), acc);
            const int n = nt * 16 + col;
            float s = 0.0f;
            #pragma unroll
            for (int j = 0; j < 8; ++j) {
                const int m = row0 + j + 8 * half;
                s += fmaxf(acc[j] + b3l[n], 0.0f) * mlds[m];
            }
            atomicAdd(&accum[n], s);
        }
    }
    __syncthreads();
    if (tid < NH) atomicAdd(&gsum[b * NH + tid], accum[tid]);
}

// ---------------------------------------------------------------------------
// Kernel 3: scoring head per complex (8 x (128x128 + 128) — trivial)
// ---------------------------------------------------------------------------
__global__ __launch_bounds__(NH) void dock_head_kernel(
    const float* __restrict__ gsum, const float* __restrict__ gcnt,
    const float* __restrict__ Wr1, const float* __restrict__ br1,
    const float* __restrict__ Wr2, const float* __restrict__ br2,
    float* __restrict__ out) {
    __shared__ float repr[NH];
    __shared__ float red[NH];
    const int bC = blockIdx.x;
    const int t = threadIdx.x;
    const float cnt = gcnt[bC];
    const float denom = fmaxf(cnt, 1.0f);
    repr[t] = gsum[bC * NH + t] / denom;
    __syncthreads();
    float acc = br1[t];
    #pragma unroll 8
    for (int k = 0; k < NH; ++k) acc = fmaf(repr[k], Wr1[k * NH + t], acc);
    red[t] = fmaxf(acc, 0.0f) * Wr2[t];
    __syncthreads();
    for (int s = NH / 2; s > 0; s >>= 1) {
        if (t < s) red[t] += red[t + s];
        __syncthreads();
    }
    if (t == 0) out[bC] = (cnt > 0.0f) ? (red[0] + br2[0]) : 0.0f;
}

// ---------------------------------------------------------------------------
// Host launcher
// ---------------------------------------------------------------------------
extern "C" void kernel_launch(void* const* d_in, const int* in_sizes, int n_in,
                              void* d_out, int out_size, void* d_ws, size_t ws_size,
                              hipStream_t stream) {
    const float* protein_pos = (const float*)d_in[0];
    const float* ligand_pos  = (const float*)d_in[1];
    const float* prot_emb    = (const float*)d_in[2];
    const float* lig_emb     = (const float*)d_in[3];
    const float* W1          = (const float*)d_in[4];
    const float* b1          = (const float*)d_in[5];
    const float* W2          = (const float*)d_in[6];
    const float* b2          = (const float*)d_in[7];
    const float* W3          = (const float*)d_in[8];
    const float* b3          = (const float*)d_in[9];
    const float* Wr1         = (const float*)d_in[10];
    const float* br1         = (const float*)d_in[11];
    const float* Wr2         = (const float*)d_in[12];
    const float* br2         = (const float*)d_in[13];
    const int*   ptype       = (const int*)d_in[14];
    const int*   ltype       = (const int*)d_in[15];
    // d_in[16], d_in[17]: batch indices — implied by contiguous layout.

    float* out = (float*)d_out;

    // workspace layout (floats): Abuf[4096*128] | Cbuf[512*128] | gsum[8*128] | gcnt[8]
    float* Abuf = (float*)d_ws;
    float* Cbuf = Abuf + (size_t)NB * NP * NH;
    float* gsum = Cbuf + (size_t)NB * NL * NH;
    float* gcnt = gsum + (size_t)NB * NH;

    dock_zero_kernel<<<4, 256, 0, stream>>>(gsum, gcnt);

    dock_precompute_kernel<<<NB * NP + NB * NL, NH, 0, stream>>>(
        prot_emb, lig_emb, ptype, ltype, W1, b1, Abuf, Cbuf);

    dock_pair_mlp_kernel<<<NB * NP / 2, 256, 0, stream>>>(
        protein_pos, ligand_pos, W1, W2, b2, W3, b3, Abuf, Cbuf, gsum, gcnt);

    dock_head_kernel<<<NB, NH, 0, stream>>>(gsum, gcnt, Wr1, br1, Wr2, br2, out);
}
